// Encoder_26852135534604
// MI455X (gfx1250) — matrix operand order repaired and checked by reading, not compiled, not run on hardware
//
#include <hip/hip_runtime.h>

// ---------------------------------------------------------------------------
// AGCRN encoder on MI455X (gfx1250, wave32, WMMA).
// ~1.1 TFLOP of matmul vs ~0.5 GB HBM traffic -> compute/L2 bound.
// Graph GEMM: LDS-staged A slab via global_load_async_to_lds_b128 (ASYNCcnt),
// 2x4 register blocking, v_wmma_f32_16x16x32_f16 everywhere.
// ---------------------------------------------------------------------------

#define N_NODES 800   // multiple of 16
#define B_      128
#define T_      12
#define OUT_    64
#define EMB     10

#define GA_ASYNC 1    // use global_load_async_to_lds_b128 for A-slab staging

typedef _Float16 half_t;
typedef _Float16 v16h __attribute__((ext_vector_type(16)));
typedef _Float16 v8h  __attribute__((ext_vector_type(8)));
typedef float    v8f  __attribute__((ext_vector_type(8)));

union V16 { v16h v; v8h h[2]; };

__device__ __forceinline__ v8f wmma_f16(v16h a, v16h b, v8f c) {
  return __builtin_amdgcn_wmma_f32_16x16x32_f16(false, a, false, b, (short)0, c,
                                                false, false);
}

// ---------------------------------------------------------------------------
// A = softmax(relu(E E^T)) row-wise, stored f16 row-major [N][N].
// ---------------------------------------------------------------------------
__global__ __launch_bounds__(256) void supports_k(const float* __restrict__ E,
                                                  half_t* __restrict__ Aad) {
  __shared__ float sE[N_NODES * EMB];
  __shared__ float row[N_NODES];
  __shared__ float red[256];
  int tid = threadIdx.x;
  int n = blockIdx.x;
  for (int i = tid; i < N_NODES * EMB; i += 256) sE[i] = E[i];
  __syncthreads();

  float lmax = -1e30f;
  for (int m = tid; m < N_NODES; m += 256) {
    float d = 0.f;
    for (int e = 0; e < EMB; e++) d += sE[n * EMB + e] * sE[m * EMB + e];
    d = fmaxf(d, 0.f);
    row[m] = d;
    lmax = fmaxf(lmax, d);
  }
  red[tid] = lmax;
  __syncthreads();
  for (int s = 128; s > 0; s >>= 1) {
    if (tid < s) red[tid] = fmaxf(red[tid], red[tid + s]);
    __syncthreads();
  }
  float mx = red[0];
  __syncthreads();
  float lsum = 0.f;
  for (int m = tid; m < N_NODES; m += 256) {
    float e = __expf(row[m] - mx);
    row[m] = e;
    lsum += e;
  }
  red[tid] = lsum;
  __syncthreads();
  for (int s = 128; s > 0; s >>= 1) {
    if (tid < s) red[tid] += red[tid + s];
    __syncthreads();
  }
  float inv = 1.f / red[0];
  for (int m = tid; m < N_NODES; m += 256)
    Aad[(size_t)n * N_NODES + m] = (half_t)(row[m] * inv);
}

// ---------------------------------------------------------------------------
// Wt[n][k][o][i] = sum_d E[n][d] * W0[d][k][i][o]  (i contiguous, zero pad).
// ---------------------------------------------------------------------------
__global__ __launch_bounds__(256) void node_weights_k(
    const float* __restrict__ W0, const float* __restrict__ E,
    half_t* __restrict__ Wt, int Ci, int Co, int CP) {
  size_t total = (size_t)N_NODES * 2 * Co * CP;
  size_t stride = (size_t)gridDim.x * blockDim.x;
  for (size_t idx = (size_t)blockIdx.x * blockDim.x + threadIdx.x; idx < total;
       idx += stride) {
    int i = (int)(idx % CP);
    size_t r = idx / CP;
    int o = (int)(r % Co); r /= Co;
    int k = (int)(r % 2);
    int n = (int)(r / 2);
    float v = 0.f;
    if (i < Ci) {
      for (int d = 0; d < EMB; d++)
        v += E[n * EMB + d] * W0[(((size_t)d * 2 + k) * Ci + i) * Co + o];
    }
    Wt[idx] = (half_t)v;
  }
}

__global__ __launch_bounds__(256) void node_bias_k(const float* __restrict__ b0,
                                                   const float* __restrict__ E,
                                                   float* __restrict__ bn,
                                                   int Co) {
  int idx = blockIdx.x * blockDim.x + threadIdx.x;
  if (idx >= N_NODES * Co) return;
  int o = idx % Co, n = idx / Co;
  float v = 0.f;
  for (int d = 0; d < EMB; d++) v += E[n * EMB + d] * b0[d * Co + o];
  bn[idx] = v;
}

// h[n][b][o] <- init_state[l][b][n][o]
__global__ __launch_bounds__(256) void init_h_k(const float* __restrict__ ist,
                                                float* __restrict__ h0,
                                                float* __restrict__ h1) {
  size_t per = (size_t)B_ * N_NODES * OUT_;
  size_t stride = (size_t)gridDim.x * blockDim.x;
  for (size_t i = (size_t)blockIdx.x * blockDim.x + threadIdx.x; i < per;
       i += stride) {
    int o = (int)(i % OUT_);
    size_t r = i / OUT_;
    int n = (int)(r % N_NODES);
    int b = (int)(r / N_NODES);
    size_t dst = ((size_t)n * B_ + b) * OUT_ + o;
    h0[dst] = ist[i];
    h1[dst] = ist[per + i];
  }
}

// ---------------------------------------------------------------------------
// Concat value for channel c of (n, b).
// ---------------------------------------------------------------------------
__device__ __forceinline__ float concat_val(const float* __restrict__ asrc,
                                            int mode, int t,
                                            const float* __restrict__ h,
                                            const float* __restrict__ zr, int n,
                                            int b, int c, int Ci_a) {
  if (c < Ci_a) {
    return (mode == 0) ? asrc[(((size_t)b * T_ + t) * N_NODES + n) * 2 + c]
                       : asrc[((size_t)n * B_ + b) * OUT_ + c];
  }
  if (c < Ci_a + OUT_) {
    int o = c - Ci_a;
    float hv = h[((size_t)n * B_ + b) * OUT_ + o];
    if (zr) hv *= zr[((size_t)n * B_ + b) * 128 + o];
    return hv;
  }
  return 0.f;
}

// Two passes, each coalesced in its destination's native order:
//   xc_rm[n][b][c]   and   xcT[(b*CP+c)][n]
__global__ __launch_bounds__(256) void concat_k(
    const float* __restrict__ asrc, int mode, int t,
    const float* __restrict__ h, const float* __restrict__ zr,
    half_t* __restrict__ xcT, half_t* __restrict__ xc_rm, int Ci_a, int CP) {
  size_t total = (size_t)N_NODES * B_ * CP;
  size_t stride = (size_t)gridDim.x * blockDim.x;
  size_t t0 = (size_t)blockIdx.x * blockDim.x + threadIdx.x;
  for (size_t idx = t0; idx < total; idx += stride) {
    int c = (int)(idx % CP);
    size_t r = idx / CP;
    int b = (int)(r % B_);
    int n = (int)(r / B_);
    xc_rm[idx] = (half_t)concat_val(asrc, mode, t, h, zr, n, b, c, Ci_a);
  }
  for (size_t idx = t0; idx < total; idx += stride) {
    int n = (int)(idx % N_NODES);
    size_t j = idx / N_NODES;
    int c = (int)(j % CP);
    int b = (int)(j / CP);
    xcT[idx] = (half_t)concat_val(asrc, mode, t, h, zr, n, b, c, Ci_a);
  }
}

// ---------------------------------------------------------------------------
// Graph conv GEMM: xg1[n][(b,c)] = sum_m A[n][m] * xcT[(b,c)][m]
// Block: 8 waves share one 32-row A slab (LDS, async-staged).
// Wave: 2 M-tiles x 4 J-tiles -> 8 acc tiles, 200 WMMA per wave.
// grid = (CP/4, 25).
// ---------------------------------------------------------------------------
__global__ __launch_bounds__(256) void gemm_graph_k(
    const half_t* __restrict__ Aad, const half_t* __restrict__ xcT,
    half_t* __restrict__ xg1, int CP) {
  __shared__ half_t sA[32 * N_NODES];  // 51,200 B
  int tid = threadIdx.x;
  int lane = tid & 31, wave = tid >> 5;
  int mg = blockIdx.y;  // M rows [mg*32, mg*32+32)
  const half_t* gA = Aad + (size_t)mg * 32 * N_NODES;

#if GA_ASYNC
  // CDNA5 async copy: per-lane LDS[vdst] <- MEM[saddr + vaddr], ASYNCcnt.
  // LDS byte address derived from the pointer (ptrtoint escapes sA so the
  // memory-clobbered asm is a visible writer; low 32 bits of a flat LDS
  // address are the LDS byte offset per the aperture rules).
  for (int idx = tid; idx < 3200; idx += 256) {
    unsigned ldsOff = (unsigned)(unsigned long long)(sA + (size_t)idx * 8);
    unsigned goff = (unsigned)(idx * 16);
    asm volatile("global_load_async_to_lds_b128 %0, %1, %2"
                 :: "v"(ldsOff), "v"(goff), "s"(gA) : "memory");
  }
  asm volatile("s_wait_asynccnt 0" ::: "memory");
  __syncthreads();
#else
  for (int idx = tid; idx < 3200; idx += 256)
    ((v8h*)sA)[idx] = ((const v8h*)gA)[idx];
  __syncthreads();
#endif

  int r16 = lane & 15, hs = lane >> 4;
  int jt0 = (blockIdx.x * 8 + wave) * 4;  // 4 consecutive J tiles
  const half_t* xcol0 = xcT + (size_t)(jt0 * 16 + r16) * N_NODES;
  const half_t* arow0 = sA + (size_t)r16 * N_NODES;
  const half_t* arow1 = sA + (size_t)(16 + r16) * N_NODES;

  v8f acc[2][4];
  const v8f vzero = {0.f, 0.f, 0.f, 0.f, 0.f, 0.f, 0.f, 0.f};
  for (int mt = 0; mt < 2; mt++)
    for (int j = 0; j < 4; j++) acc[mt][j] = vzero;

  for (int k0 = 0; k0 < N_NODES; k0 += 32) {
    V16 a0, a1, bv[4];
    // issue all loads first (clause-able), then all WMMAs
    a0.h[0] = *(const v8h*)(arow0 + k0 + hs * 8);
    a0.h[1] = *(const v8h*)(arow0 + k0 + 16 + hs * 8);
    a1.h[0] = *(const v8h*)(arow1 + k0 + hs * 8);
    a1.h[1] = *(const v8h*)(arow1 + k0 + 16 + hs * 8);
#pragma unroll
    for (int j = 0; j < 4; j++) {
      const half_t* xc = xcol0 + (size_t)j * 16 * N_NODES;
      bv[j].h[0] = *(const v8h*)(xc + k0 + hs * 8);
      bv[j].h[1] = *(const v8h*)(xc + k0 + 16 + hs * 8);
    }
#pragma unroll
    for (int j = 0; j < 4; j++) {
      acc[0][j] = wmma_f16(a0.v, bv[j].v, acc[0][j]);
      acc[1][j] = wmma_f16(a1.v, bv[j].v, acc[1][j]);
    }
  }
#pragma unroll
  for (int mt = 0; mt < 2; mt++)
#pragma unroll
    for (int j = 0; j < 4; j++) {
      int col = (jt0 + j) * 16 + r16;
      int b = col / CP, c = col % CP;
      for (int r = 0; r < 8; r++) {  // D: M = r + 8*hs
        int m = mg * 32 + mt * 16 + r + 8 * hs;
        xg1[((size_t)m * B_ + b) * CP + c] = (half_t)acc[mt][j][r];
      }
    }
}

// ---------------------------------------------------------------------------
// Per-node GEMM core, 2x2 tiles: acc[mt][ot] over b0+mt*16 / o0+ot*16.
// ---------------------------------------------------------------------------
__device__ __forceinline__ void pernode_mm2(const half_t* __restrict__ xc_rm,
                                            const half_t* __restrict__ xg1,
                                            const half_t* __restrict__ Wt,
                                            int n, int b0, int o0, int CP,
                                            int Co, int r16, int hs,
                                            v8f acc[2][2]) {
  for (int kk = 0; kk < 2; kk++) {
    const half_t* act = (kk == 0 ? xc_rm : xg1) + (size_t)n * B_ * CP;
    const half_t* ar0 = act + (size_t)(b0 + r16) * CP;
    const half_t* ar1 = act + (size_t)(b0 + 16 + r16) * CP;
    const half_t* wr0 =
        Wt + (((size_t)n * 2 + kk) * Co + (o0 + r16)) * (size_t)CP;
    const half_t* wr1 = wr0 + (size_t)16 * CP;
    for (int i0 = 0; i0 < CP; i0 += 32) {
      V16 a0, a1, b0v, b1v;
      a0.h[0] = *(const v8h*)(ar0 + i0 + hs * 8);
      a0.h[1] = *(const v8h*)(ar0 + i0 + 16 + hs * 8);
      a1.h[0] = *(const v8h*)(ar1 + i0 + hs * 8);
      a1.h[1] = *(const v8h*)(ar1 + i0 + 16 + hs * 8);
      b0v.h[0] = *(const v8h*)(wr0 + i0 + hs * 8);
      b0v.h[1] = *(const v8h*)(wr0 + i0 + 16 + hs * 8);
      b1v.h[0] = *(const v8h*)(wr1 + i0 + hs * 8);
      b1v.h[1] = *(const v8h*)(wr1 + i0 + 16 + hs * 8);
      acc[0][0] = wmma_f16(a0.v, b0v.v, acc[0][0]);
      acc[0][1] = wmma_f16(a0.v, b1v.v, acc[0][1]);
      acc[1][0] = wmma_f16(a1.v, b0v.v, acc[1][0]);
      acc[1][1] = wmma_f16(a1.v, b1v.v, acc[1][1]);
    }
  }
}

// Gates: zr[n][b][0:128] = sigmoid(gemm + bias). grid = (800, 2).
__global__ __launch_bounds__(256) void pernode_gates_k(
    const half_t* __restrict__ xc_rm, const half_t* __restrict__ xg1,
    const half_t* __restrict__ Wt, const float* __restrict__ bias,
    float* __restrict__ zr, int CP) {
  const int Co = 128;
  int lane = threadIdx.x & 31, wave = threadIdx.x >> 5;
  int n = blockIdx.x;
  int tile = blockIdx.y * 8 + wave;  // 16 tiles: 4 bgroups x 4 ogroups
  int og = tile & 3, bg = tile >> 2;
  int r16 = lane & 15, hs = lane >> 4;
  int b0 = bg * 32, o0 = og * 32;
  v8f acc[2][2];
  const v8f vzero = {0.f, 0.f, 0.f, 0.f, 0.f, 0.f, 0.f, 0.f};
  acc[0][0] = vzero; acc[0][1] = vzero; acc[1][0] = vzero; acc[1][1] = vzero;
  pernode_mm2(xc_rm, xg1, Wt, n, b0, o0, CP, Co, r16, hs, acc);
#pragma unroll
  for (int ot = 0; ot < 2; ot++) {
    int o = o0 + ot * 16 + r16;
    float bv = bias[n * Co + o];
#pragma unroll
    for (int mt = 0; mt < 2; mt++)
      for (int r = 0; r < 8; r++) {
        int b = b0 + mt * 16 + r + 8 * hs;
        float v = acc[mt][ot][r] + bv;
        zr[((size_t)n * B_ + b) * 128 + o] = 1.f / (1.f + __expf(-v));
      }
  }
}

// Candidate + GRU state update. grid = (800, 1).
__global__ __launch_bounds__(256) void pernode_cand_k(
    const half_t* __restrict__ xc_rm, const half_t* __restrict__ xg1,
    const half_t* __restrict__ Wt, const float* __restrict__ bias,
    const float* __restrict__ zr, float* __restrict__ h,
    float* __restrict__ outp, int t, int CP) {
  const int Co = 64;
  int lane = threadIdx.x & 31, wave = threadIdx.x >> 5;
  int n = blockIdx.x;
  int tile = wave;  // 8 tiles: 4 bgroups x 2 ogroups
  int og = tile & 1, bg = tile >> 1;
  int r16 = lane & 15, hs = lane >> 4;
  int b0 = bg * 32, o0 = og * 32;
  v8f acc[2][2];
  const v8f vzero = {0.f, 0.f, 0.f, 0.f, 0.f, 0.f, 0.f, 0.f};
  acc[0][0] = vzero; acc[0][1] = vzero; acc[1][0] = vzero; acc[1][1] = vzero;
  pernode_mm2(xc_rm, xg1, Wt, n, b0, o0, CP, Co, r16, hs, acc);
#pragma unroll
  for (int ot = 0; ot < 2; ot++) {
    int o = o0 + ot * 16 + r16;
    float bv = bias[n * Co + o];
#pragma unroll
    for (int mt = 0; mt < 2; mt++)
      for (int r = 0; r < 8; r++) {
        int b = b0 + mt * 16 + r + 8 * hs;
        float hc = tanhf(acc[mt][ot][r] + bv);
        float rg = zr[((size_t)n * B_ + b) * 128 + 64 + o];
        size_t hidx = ((size_t)n * B_ + b) * OUT_ + o;
        float hn = rg * h[hidx] + (1.f - rg) * hc;
        h[hidx] = hn;
        if (outp) outp[(((size_t)b * T_ + t) * N_NODES + n) * OUT_ + o] = hn;
      }
  }
}

// lasts[l][b][n][o] <- h_l[n][b][o]
__global__ __launch_bounds__(256) void lasts_k(const float* __restrict__ h0,
                                               const float* __restrict__ h1,
                                               float* __restrict__ lasts) {
  size_t per = (size_t)B_ * N_NODES * OUT_;
  size_t stride = (size_t)gridDim.x * blockDim.x;
  for (size_t i = (size_t)blockIdx.x * blockDim.x + threadIdx.x; i < per;
       i += stride) {
    int o = (int)(i % OUT_);
    size_t r = i / OUT_;
    int n = (int)(r % N_NODES);
    int b = (int)(r / N_NODES);
    size_t src = ((size_t)n * B_ + b) * OUT_ + o;
    lasts[i] = h0[src];
    lasts[per + i] = h1[src];
  }
}

// ---------------------------------------------------------------------------
extern "C" void kernel_launch(void* const* d_in, const int* in_sizes, int n_in,
                              void* d_out, int out_size, void* d_ws,
                              size_t ws_size, hipStream_t stream) {
  const float* x   = (const float*)d_in[0];
  const float* ist = (const float*)d_in[1];
  const float* E   = (const float*)d_in[2];
  const float* Wg0 = (const float*)d_in[3];
  const float* bg0 = (const float*)d_in[4];
  const float* Wc0 = (const float*)d_in[5];
  const float* bc0 = (const float*)d_in[6];
  const float* Wg1 = (const float*)d_in[7];
  const float* bg1 = (const float*)d_in[8];
  const float* Wc1 = (const float*)d_in[9];
  const float* bc1 = (const float*)d_in[10];
  float* out = (float*)d_out;
  float* lasts = out + (size_t)B_ * T_ * N_NODES * OUT_;

  const int CP0 = 96;    // layer0 Ci=66 padded to 3*32
  const int CP1 = 128;   // layer1 Ci=128

  char* ws = (char*)d_ws;
  size_t off = 0;
  auto carve = [&](size_t bytes) -> char* {
    char* p = ws + off;
    off = (off + bytes + 255) & ~(size_t)255;
    return p;
  };
  half_t* Aad  = (half_t*)carve((size_t)N_NODES * N_NODES * 2);
  half_t* Wtg0 = (half_t*)carve((size_t)N_NODES * 2 * 128 * CP0 * 2);
  half_t* Wtc0 = (half_t*)carve((size_t)N_NODES * 2 * 64 * CP0 * 2);
  half_t* Wtg1 = (half_t*)carve((size_t)N_NODES * 2 * 128 * CP1 * 2);
  half_t* Wtc1 = (half_t*)carve((size_t)N_NODES * 2 * 64 * CP1 * 2);
  float* bng0 = (float*)carve((size_t)N_NODES * 128 * 4);
  float* bnc0 = (float*)carve((size_t)N_NODES * 64 * 4);
  float* bng1 = (float*)carve((size_t)N_NODES * 128 * 4);
  float* bnc1 = (float*)carve((size_t)N_NODES * 64 * 4);
  half_t* xcT   = (half_t*)carve((size_t)B_ * 128 * N_NODES * 2);
  half_t* xc_rm = (half_t*)carve((size_t)N_NODES * B_ * 128 * 2);
  half_t* xg1   = (half_t*)carve((size_t)N_NODES * B_ * 128 * 2);
  float* zr = (float*)carve((size_t)N_NODES * B_ * 128 * 4);
  float* h0 = (float*)carve((size_t)N_NODES * B_ * OUT_ * 4);
  float* h1 = (float*)carve((size_t)N_NODES * B_ * OUT_ * 4);
  (void)in_sizes; (void)n_in; (void)out_size; (void)ws_size;

  // ---- per-launch precompute ----
  supports_k<<<N_NODES, 256, 0, stream>>>(E, Aad);
  node_weights_k<<<4096, 256, 0, stream>>>(Wg0, E, Wtg0, 66, 128, CP0);
  node_weights_k<<<4096, 256, 0, stream>>>(Wc0, E, Wtc0, 66, 64, CP0);
  node_weights_k<<<4096, 256, 0, stream>>>(Wg1, E, Wtg1, 128, 128, CP1);
  node_weights_k<<<4096, 256, 0, stream>>>(Wc1, E, Wtc1, 128, 64, CP1);
  node_bias_k<<<(N_NODES * 128 + 255) / 256, 256, 0, stream>>>(bg0, E, bng0, 128);
  node_bias_k<<<(N_NODES * 64 + 255) / 256, 256, 0, stream>>>(bc0, E, bnc0, 64);
  node_bias_k<<<(N_NODES * 128 + 255) / 256, 256, 0, stream>>>(bg1, E, bng1, 128);
  node_bias_k<<<(N_NODES * 64 + 255) / 256, 256, 0, stream>>>(bc1, E, bnc1, 64);
  init_h_k<<<4096, 256, 0, stream>>>(ist, h0, h1);

  dim3 blk(256);
  for (int t = 0; t < T_; t++) {
    // ---------- layer 0 (input x_t, state h0) ----------
    concat_k<<<8192, 256, 0, stream>>>(x, 0, t, h0, (const float*)nullptr,
                                       xcT, xc_rm, 2, CP0);
    gemm_graph_k<<<dim3(CP0 / 4, 25), blk, 0, stream>>>(Aad, xcT, xg1, CP0);
    pernode_gates_k<<<dim3(N_NODES, 2), blk, 0, stream>>>(xc_rm, xg1, Wtg0,
                                                          bng0, zr, CP0);
    concat_k<<<8192, 256, 0, stream>>>(x, 0, t, h0, zr, xcT, xc_rm, 2, CP0);
    gemm_graph_k<<<dim3(CP0 / 4, 25), blk, 0, stream>>>(Aad, xcT, xg1, CP0);
    pernode_cand_k<<<dim3(N_NODES, 1), blk, 0, stream>>>(
        xc_rm, xg1, Wtc0, bnc0, zr, h0, (float*)nullptr, t, CP0);
    // ---------- layer 1 (input h0, state h1) ----------
    concat_k<<<8192, 256, 0, stream>>>(h0, 1, t, h1, (const float*)nullptr,
                                       xcT, xc_rm, OUT_, CP1);
    gemm_graph_k<<<dim3(CP1 / 4, 25), blk, 0, stream>>>(Aad, xcT, xg1, CP1);
    pernode_gates_k<<<dim3(N_NODES, 2), blk, 0, stream>>>(xc_rm, xg1, Wtg1,
                                                          bng1, zr, CP1);
    concat_k<<<8192, 256, 0, stream>>>(h0, 1, t, h1, zr, xcT, xc_rm, OUT_, CP1);
    gemm_graph_k<<<dim3(CP1 / 4, 25), blk, 0, stream>>>(Aad, xcT, xg1, CP1);
    pernode_cand_k<<<dim3(N_NODES, 1), blk, 0, stream>>>(
        xc_rm, xg1, Wtc1, bnc1, zr, h1, out, t, CP1);
  }
  lasts_k<<<4096, 256, 0, stream>>>(h0, h1, lasts);
}
